// SelfAttention_86535001079843
// MI455X (gfx1250) — matrix operand order repaired and checked by reading, not compiled, not run on hardware
//
#include <hip/hip_runtime.h>
#include <hip/hip_bf16.h>

// ---------------------------------------------------------------------------
// Bilinear self-attention for MI455X (gfx1250), wave32 + WMMA bf16.
//   x          : [B=8, 256, 1024]  f32
//   innerprod  : [H=8, 256, 256]   f32
//   tovalue    : [H=8, 256, 256]   f32
//   unifyheads : [H=8, 256, 256]   f32
//   out        : [B, 256, 1024]    f32
// Score scale (1/sqrt(256)) AND log2(e) are pre-folded into the xa
// projection, so softmax runs natively on v_exp_f32 (exp2) with no
// per-element pre-multiply.
// ---------------------------------------------------------------------------

#define NB    8
#define NH    8
#define NBH   64
#define DIMC  256
#define LSEQ  1024
// (1/sqrt(256)) * log2(e): scores leave k_proj<0> already in exp2 domain.
#define SSCALE 0.09016844f

typedef __bf16 bf16_t;
typedef __attribute__((ext_vector_type(16))) __bf16 v16bf;
typedef __attribute__((ext_vector_type(8)))  __bf16 v8bf;
typedef __attribute__((ext_vector_type(8)))  float  v8f;

static __device__ inline bf16_t f2bf(float f) {
    unsigned u = __builtin_bit_cast(unsigned, f);
    unsigned r = u + 0x7FFFu + ((u >> 16) & 1u);     // round-to-nearest-even
    unsigned short h = (unsigned short)(r >> 16);
    return __builtin_bit_cast(bf16_t, h);
}

static __device__ inline float fexp2(float f) { return __builtin_amdgcn_exp2f(f); }

static __device__ inline v8f vzero8() {
    v8f z;
    for (int e = 0; e < 8; ++e) z[e] = 0.0f;
    return z;
}

// WMMA bf16 16x16x32, fp32 accumulate.
static __device__ inline v8f wmma_bf16(v16bf a, v16bf b, v8f c) {
    return __builtin_amdgcn_wmma_f32_16x16x32_bf16(
        /*neg_a=*/false, a, /*neg_b=*/false, b,
        /*c_mod=*/(short)0, c, /*reuse_a=*/false, /*reuse_b=*/false);
}

// A-fragment (16x32, MxK) from row-major [M][K] LDS.
// ISA: lanes 0-15 row M=lane, K in {0..7}U{16..23}; lanes 16-31 row M=lane-16,
// K in {8..15}U{24..31}  -> two contiguous 16B runs per lane.
static __device__ inline v16bf load_fragA(const bf16_t* base, int row0, int ld, int k0) {
    const int lane = threadIdx.x & 31;
    const int half = lane >> 4;
    const int idx  = lane & 15;
    const bf16_t* p = base + (row0 + idx) * ld + k0 + half * 8;
    union { v16bf v; v8bf h[2]; } u;
    u.h[0] = *reinterpret_cast<const v8bf*>(p);        // K run0
    u.h[1] = *reinterpret_cast<const v8bf*>(p + 16);   // K run1 (+16 elements)
    return u.v;
}

// B-fragment (32x16, KxN) from an [N][K] (column-major B) LDS layout.
// Lane holds column N=lane&15, K in {0..7}U{16..23} (lanes 0-15) or
// {8..15}U{24..31} (lanes 16-31) -> two 16B runs per lane, matching fragA.
static __device__ inline v16bf load_fragB(const bf16_t* base, int col0, int ld, int k0) {
    const int lane = threadIdx.x & 31;
    const int half = lane >> 4;
    const int idx  = lane & 15;
    const bf16_t* p = base + (col0 + idx) * ld + k0 + half * 8;
    union { v16bf v; v8bf h[2]; } u;
    u.h[0] = *reinterpret_cast<const v8bf*>(p);
    u.h[1] = *reinterpret_cast<const v8bf*>(p + 16);
    return u.v;
}

union Pack8 { v8bf v; uint4 u; };

// ---------------------------------------------------------------------------
// Prep 1: xT[b][l][k] = bf16(x[b][k][l])
// ---------------------------------------------------------------------------
__global__ void k_xT(const float* __restrict__ x, bf16_t* __restrict__ xT) {
    int tid = blockIdx.x * blockDim.x + threadIdx.x;
    const int total = NB * DIMC * LSEQ;
    for (int i = tid; i < total; i += gridDim.x * blockDim.x) {
        int b   = i / (DIMC * LSEQ);
        int rem = i - b * (DIMC * LSEQ);
        int l   = rem >> 8;
        int k   = rem & 255;
        xT[i] = f2bf(x[(b * DIMC + k) * LSEQ + l]);
    }
}

// ---------------------------------------------------------------------------
// Prep 2: weight conversion. TRANS=1: dst[h][q][k]=src[h][k][q], else copy.
// ---------------------------------------------------------------------------
template <bool TRANS>
__global__ void k_prepW(const float* __restrict__ src, bf16_t* __restrict__ dst) {
    int tid = blockIdx.x * blockDim.x + threadIdx.x;
    const int total = NH * DIMC * DIMC;
    for (int i = tid; i < total; i += gridDim.x * blockDim.x) {
        if (TRANS) {
            int h = i >> 16, q = (i >> 8) & 255, k = i & 255;
            dst[i] = f2bf(src[h * 65536 + k * 256 + q]);
        } else {
            dst[i] = f2bf(src[i]);
        }
    }
}

// ---------------------------------------------------------------------------
// Projection GEMM per (b,h): out[m][i] = sum_k Wbf[h][m][k] * xT[b][i][k]
//   MODE=0: store SSCALE * result as xaT[bh][i][q]  (scale+log2e pre-folded)
//   MODE=1: store v[bh][t][i]
// Block = (bh, i-tile of 64); 8 waves cover M=256 (32 rows each) x N=64.
// ---------------------------------------------------------------------------
template <int MODE>
__global__ void k_proj(const bf16_t* __restrict__ Wbf, const bf16_t* __restrict__ xT,
                       bf16_t* __restrict__ out) {
    __shared__ __align__(16) bf16_t WT[DIMC * 32];  // [m][k]  16KB
    __shared__ __align__(16) bf16_t BT[64 * 32];    // [i][k]   4KB

    const int bh = blockIdx.x >> 4;
    const int i0 = (blockIdx.x & 15) * 64;
    const int b  = bh >> 3;
    const int h  = bh & 7;

    const int wave = threadIdx.x >> 5;
    const int lane = threadIdx.x & 31;
    const int half = lane >> 4;
    const int nidx = lane & 15;

    v8f acc[8];
    for (int t = 0; t < 8; ++t) acc[t] = vzero8();

#pragma unroll 1
    for (int k0 = 0; k0 < DIMC; k0 += 32) {
        // WT: 256 rows x 32 k  (1024 uint4, 4/thread), contiguous per row
        for (int c = threadIdx.x; c < 1024; c += 256) {
            int row = c >> 2, part = c & 3;
            reinterpret_cast<uint4*>(WT)[c] = *reinterpret_cast<const uint4*>(
                &Wbf[h * 65536 + row * 256 + k0 + part * 8]);
        }
        // BT: 64 rows x 32 k from xT[b][i0+row][k0..] (256 uint4, 1/thread)
        for (int c = threadIdx.x; c < 256; c += 256) {
            int row = c >> 2, part = c & 3;
            reinterpret_cast<uint4*>(BT)[c] = *reinterpret_cast<const uint4*>(
                &xT[(b * LSEQ + i0 + row) * DIMC + k0 + part * 8]);
        }
        __syncthreads();

        v16bf afr[2], bfr[4];
        for (int a = 0; a < 2; ++a) afr[a] = load_fragA(WT, 32 * wave + 16 * a, 32, 0);
        for (int bb = 0; bb < 4; ++bb) bfr[bb] = load_fragB(BT, 16 * bb, 32, 0);
        for (int a = 0; a < 2; ++a)
            for (int bb = 0; bb < 4; ++bb)
                acc[a * 4 + bb] = wmma_bf16(afr[a], bfr[bb], acc[a * 4 + bb]);
        __syncthreads();
    }

    for (int a = 0; a < 2; ++a) {
        for (int bb = 0; bb < 4; ++bb) {
            int m_base = 32 * wave + 16 * a + 8 * half;
            int i      = i0 + 16 * bb + nidx;
            if (MODE == 0) {  // xaT[bh][i][q] * SSCALE: 8 consecutive q -> 16B store
                Pack8 pk;
                for (int r = 0; r < 8; ++r) pk.v[r] = f2bf(acc[a * 4 + bb][r] * SSCALE);
                *reinterpret_cast<uint4*>(&out[(bh * LSEQ + i) * DIMC + m_base]) = pk.u;
            } else {          // v[bh][t][i]
                for (int r = 0; r < 8; ++r)
                    out[(bh * DIMC + m_base + r) * LSEQ + i] = f2bf(acc[a * 4 + bb][r]);
            }
        }
    }
}

// ---------------------------------------------------------------------------
// Softmax row stats over j for s[i][j] = (xa^T x)[i][j]  (exp2 domain).
// Block = (bh, i-tile of 64). (wave&3) -> 16 i-rows, (wave>>2) -> 16-col half
// of each 32-wide j step. Lane-LOCAL online max/sum-exp2 over the j columns a
// lane owns; the 16-lane butterfly combine happens ONCE at the end.
// Outputs m and 1/l (both in exp2 domain).
// ---------------------------------------------------------------------------
__global__ void k_stats(const bf16_t* __restrict__ xaT, const bf16_t* __restrict__ xT,
                        float* __restrict__ mOut, float* __restrict__ lInvOut) {
    __shared__ __align__(16) bf16_t xaS[64 * DIMC];  // [i][q] 32KB
    __shared__ __align__(16) bf16_t xS[32 * DIMC];   // [j][q] 16KB
    __shared__ float mPart[2][64];
    __shared__ float lPart[2][64];

    const int bh = blockIdx.x >> 4;
    const int i0 = (blockIdx.x & 15) * 64;
    const int b  = bh >> 3;

    const int wave = threadIdx.x >> 5;
    const int lane = threadIdx.x & 31;
    const int half = lane >> 4;
    const int ig   = wave & 3;   // i-row group (16 rows)
    const int jh   = wave >> 2;  // j column half (16 cols)

    {
        const uint4* src = reinterpret_cast<const uint4*>(xaT + (bh * LSEQ + i0) * DIMC);
        uint4* dst = reinterpret_cast<uint4*>(xaS);
        for (int c = threadIdx.x; c < 64 * DIMC / 8; c += 256) dst[c] = src[c];
    }
    __syncthreads();

    float m_run[8], l_run[8];
    for (int r = 0; r < 8; ++r) { m_run[r] = -3.0e38f; l_run[r] = 0.0f; }

#pragma unroll 1
    for (int j0 = 0; j0 < LSEQ; j0 += 32) {
        const uint4* src = reinterpret_cast<const uint4*>(xT + (b * LSEQ + j0) * DIMC);
        uint4* dst = reinterpret_cast<uint4*>(xS);
        for (int c = threadIdx.x; c < 32 * DIMC / 8; c += 256) dst[c] = src[c];
        __syncthreads();

        v8f sc = vzero8();
#pragma unroll 2
        for (int k0 = 0; k0 < DIMC; k0 += 32) {
            v16bf a  = load_fragA(xaS, 16 * ig, DIMC, k0);
            v16bf bb = load_fragB(xS, 16 * jh, DIMC, k0);
            sc = wmma_bf16(a, bb, sc);
        }
        // Lane-local online softmax update (no cross-lane traffic here).
        for (int r = 0; r < 8; ++r) {
            float s  = sc[r];
            float mn = fmaxf(m_run[r], s);
            l_run[r] = l_run[r] * fexp2(m_run[r] - mn) + fexp2(s - mn);
            m_run[r] = mn;
        }
        __syncthreads();
    }

    // Final 16-lane combine per row: m = max(m_lane); l = sum l_lane*2^(m_lane-m).
    for (int r = 0; r < 8; ++r) {
        float mm = m_run[r];
        for (int mk = 1; mk < 16; mk <<= 1) mm = fmaxf(mm, __shfl_xor(mm, mk, 32));
        float ls = l_run[r] * fexp2(m_run[r] - mm);
        for (int mk = 1; mk < 16; mk <<= 1) ls += __shfl_xor(ls, mk, 32);
        m_run[r] = mm;
        l_run[r] = ls;
    }

    if ((lane & 15) == 0) {
        for (int r = 0; r < 8; ++r) {
            int rl = 16 * ig + 8 * half + r;
            mPart[jh][rl] = m_run[r];
            lPart[jh][rl] = l_run[r];
        }
    }
    __syncthreads();
    if (threadIdx.x < 64) {
        int rl = threadIdx.x;
        float m0 = mPart[0][rl], m1 = mPart[1][rl];
        float mm = fmaxf(m0, m1);
        float ll = lPart[0][rl] * fexp2(m0 - mm) + lPart[1][rl] * fexp2(m1 - mm);
        mOut[bh * LSEQ + i0 + rl]    = mm;
        lInvOut[bh * LSEQ + i0 + rl] = 1.0f / ll;
    }
}

// ---------------------------------------------------------------------------
// Fused attention apply. Block = (bh, j-tile of 64).
// Per i-step(32): recompute S tile (32x64) with WMMA, p = 2^(s-m)*linv -> bf16
// in LDS [j][i], then y[t][j] += v[t][i-step] @ p. y (256x64 f32) in VGPRs.
// `buf` is overlaid: xa tile in the S phase, v tile in the y phase.
// ---------------------------------------------------------------------------
__global__ void k_attn(const bf16_t* __restrict__ xaT, const bf16_t* __restrict__ xT,
                       const bf16_t* __restrict__ vG, const float* __restrict__ mS,
                       const float* __restrict__ lInvS, bf16_t* __restrict__ yT) {
    __shared__ __align__(16) bf16_t xS[64 * DIMC];  // [j][q] 32KB, persistent
    __shared__ __align__(16) bf16_t buf[32 * DIMC]; // xa [i][q] OR v [t][i] 16KB
    __shared__ __align__(16) bf16_t pS[64 * 32];    // [j][i] 4KB
    __shared__ float mlS[2 * LSEQ];                 // m | 1/l  8KB

    const int bh = blockIdx.x >> 4;
    const int j0 = (blockIdx.x & 15) * 64;
    const int b  = bh >> 3;

    const int wave = threadIdx.x >> 5;
    const int lane = threadIdx.x & 31;
    const int half = lane >> 4;
    const int nidx = lane & 15;
    const int ti = wave >> 2;  // S-phase: i sub-tile (0..1)
    const int tj = wave & 3;   // S-phase: j sub-tile (0..3)

    {   // persistent x tile [j][q] + softmax stats
        const uint4* src = reinterpret_cast<const uint4*>(xT + (b * LSEQ + j0) * DIMC);
        uint4* dst = reinterpret_cast<uint4*>(xS);
        for (int c = threadIdx.x; c < 64 * DIMC / 8; c += 256) dst[c] = src[c];
        for (int c = threadIdx.x; c < LSEQ; c += 256) {
            mlS[c]        = mS[bh * LSEQ + c];
            mlS[LSEQ + c] = lInvS[bh * LSEQ + c];
        }
    }

    v8f acc[8];
    for (int t = 0; t < 8; ++t) acc[t] = vzero8();

#pragma unroll 1
    for (int ii = 0; ii < LSEQ; ii += 32) {
        {   // buf <- xa tile [i][q]
            const uint4* src = reinterpret_cast<const uint4*>(xaT + (bh * LSEQ + ii) * DIMC);
            uint4* dst = reinterpret_cast<uint4*>(buf);
            for (int c = threadIdx.x; c < 32 * DIMC / 8; c += 256) dst[c] = src[c];
        }
        __syncthreads();

        // ---- S phase: each wave one 16x16 tile of the 32x64 score tile ----
        v8f sc = vzero8();
#pragma unroll 2
        for (int k0 = 0; k0 < DIMC; k0 += 32) {
            v16bf a  = load_fragA(buf, 16 * ti, DIMC, k0);
            v16bf bb = load_fragB(xS, 16 * tj, DIMC, k0);
            sc = wmma_bf16(a, bb, sc);
        }
        {   // p = 2^(s - m)*linv, write pS[j][i] (8 consecutive i per lane)
            Pack8 pk;
            int ibase = ii + 16 * ti + 8 * half;
            for (int r = 0; r < 8; ++r) {
                float m    = mlS[ibase + r];
                float linv = mlS[LSEQ + ibase + r];
                pk.v[r] = f2bf(fexp2(sc[r] - m) * linv);
            }
            *reinterpret_cast<uint4*>(&pS[(16 * tj + nidx) * 32 + 16 * ti + 8 * half]) = pk.u;
        }
        __syncthreads();

        {   // buf <- v tile [t][i-step]  (overlay)
            for (int c = threadIdx.x; c < 1024; c += 256) {
                int row = c >> 2, part = c & 3;
                reinterpret_cast<uint4*>(buf)[c] = *reinterpret_cast<const uint4*>(
                    &vG[(bh * DIMC + row) * LSEQ + ii + part * 8]);
            }
        }
        __syncthreads();

        // ---- y phase: wave owns t-rows [32*wave, +32), all 64 j ----
        v16bf av[2], bv[4];
        for (int a = 0; a < 2; ++a) av[a] = load_fragA(buf, 32 * wave + 16 * a, 32, 0);
        for (int bb = 0; bb < 4; ++bb) bv[bb] = load_fragB(pS, 16 * bb, 32, 0);
        for (int a = 0; a < 2; ++a)
            for (int bb = 0; bb < 4; ++bb)
                acc[a * 4 + bb] = wmma_bf16(av[a], bv[bb], acc[a * 4 + bb]);
        __syncthreads();
    }

    // store yT[bh][j][t] (8 consecutive t -> one 16B store)
    for (int a = 0; a < 2; ++a) {
        for (int bb = 0; bb < 4; ++bb) {
            int t_base = 32 * wave + 16 * a + 8 * half;
            int j      = j0 + 16 * bb + nidx;
            Pack8 pk;
            for (int r = 0; r < 8; ++r) pk.v[r] = f2bf(acc[a * 4 + bb][r]);
            *reinterpret_cast<uint4*>(&yT[(bh * LSEQ + j) * DIMC + t_base]) = pk.u;
        }
    }
}

// ---------------------------------------------------------------------------
// Unify: out[b][c][l] = sum_{h,t} Ubf[h][c][t] * yT[b*8+h][l][t]   (K = 2048)
// Block = (b, l-tile of 64); 8 waves cover M=c=256 x N=64.
// ---------------------------------------------------------------------------
__global__ void k_unify(const bf16_t* __restrict__ Ubf, const bf16_t* __restrict__ yT,
                        float* __restrict__ out) {
    __shared__ __align__(16) bf16_t US[DIMC * 32];  // [c][t] 16KB
    __shared__ __align__(16) bf16_t yS[64 * 32];    // [l][t]  4KB

    const int b  = blockIdx.x >> 4;
    const int l0 = (blockIdx.x & 15) * 64;

    const int wave = threadIdx.x >> 5;
    const int lane = threadIdx.x & 31;
    const int half = lane >> 4;
    const int nidx = lane & 15;

    v8f acc[8];
    for (int t = 0; t < 8; ++t) acc[t] = vzero8();

#pragma unroll 1
    for (int hk = 0; hk < NH * 8; ++hk) {
        const int h  = hk >> 3;
        const int t0 = (hk & 7) * 32;
        for (int c = threadIdx.x; c < 1024; c += 256) {
            int row = c >> 2, part = c & 3;
            reinterpret_cast<uint4*>(US)[c] = *reinterpret_cast<const uint4*>(
                &Ubf[h * 65536 + row * 256 + t0 + part * 8]);
        }
        for (int c = threadIdx.x; c < 256; c += 256) {
            int row = c >> 2, part = c & 3;
            reinterpret_cast<uint4*>(yS)[c] = *reinterpret_cast<const uint4*>(
                &yT[((b * NH + h) * LSEQ + l0 + row) * DIMC + t0 + part * 8]);
        }
        __syncthreads();

        v16bf afr[2], bfr[4];
        for (int a = 0; a < 2; ++a) afr[a] = load_fragA(US, 32 * wave + 16 * a, 32, 0);
        for (int bb = 0; bb < 4; ++bb) bfr[bb] = load_fragB(yS, 16 * bb, 32, 0);
        for (int a = 0; a < 2; ++a)
            for (int bb = 0; bb < 4; ++bb)
                acc[a * 4 + bb] = wmma_bf16(afr[a], bfr[bb], acc[a * 4 + bb]);
        __syncthreads();
    }

    for (int a = 0; a < 2; ++a) {
        for (int bb = 0; bb < 4; ++bb) {
            int c_base = 32 * wave + 16 * a + 8 * half;
            int l      = l0 + 16 * bb + nidx;
            for (int r = 0; r < 8; ++r)
                out[(b * DIMC + c_base + r) * LSEQ + l] = acc[a * 4 + bb][r];
        }
    }
}

// ---------------------------------------------------------------------------
// Host launcher
// ---------------------------------------------------------------------------
extern "C" void kernel_launch(void* const* d_in, const int* in_sizes, int n_in,
                              void* d_out, int out_size, void* d_ws, size_t ws_size,
                              hipStream_t stream) {
    (void)in_sizes; (void)n_in; (void)out_size; (void)ws_size;
    const float* x  = (const float*)d_in[0];
    const float* ip = (const float*)d_in[1];
    const float* tv = (const float*)d_in[2];
    const float* uh = (const float*)d_in[3];
    float* out = (float*)d_out;

    char* ws = (char*)d_ws;
    // workspace layout (all 16B aligned)
    bf16_t* xT   = (bf16_t*)(ws + 0);                        //  4 MB: [b][l][k]
    bf16_t* xaT  = (bf16_t*)(ws + (4u << 20));               // 32 MB: [bh][i][q]
    bf16_t* vG   = (bf16_t*)(ws + (36u << 20));              // 32 MB: [bh][t][i]
    bf16_t* yT   = (bf16_t*)(ws + (68u << 20));              // 32 MB: [bh][l][t]
    float*  mS   = (float*)(ws + (100u << 20));              // 256 KB
    float*  lInv = (float*)(ws + (100u << 20) + (256u << 10)); // 256 KB
    bf16_t* ipT  = (bf16_t*)(ws + (100u << 20) + (512u << 10)); // 1 MB [h][q][k]
    bf16_t* tvB  = (bf16_t*)(ws + (101u << 20) + (512u << 10)); // 1 MB [h][t][k]
    bf16_t* uhB  = (bf16_t*)(ws + (102u << 20) + (512u << 10)); // 1 MB [h][c][t]

    k_xT          <<<2048, 256, 0, stream>>>(x, xT);
    k_prepW<true> <<<256, 256, 0, stream>>>(ip, ipT);
    k_prepW<false><<<256, 256, 0, stream>>>(tv, tvB);
    k_prepW<false><<<256, 256, 0, stream>>>(uh, uhB);
    k_proj<0>     <<<NBH * 16, 256, 0, stream>>>(ipT, xT, xaT);  // xa (exp2-scaled)
    k_proj<1>     <<<NBH * 16, 256, 0, stream>>>(tvB, xT, vG);   // v
    k_stats       <<<NBH * 16, 256, 0, stream>>>(xaT, xT, mS, lInv);
    k_attn        <<<NBH * 16, 256, 0, stream>>>(xaT, xT, vG, mS, lInv, yT);
    k_unify       <<<NB * 16, 256, 0, stream>>>(uhB, yT, out);
}
